// TriangleMultiplicativeUpdate_670014898471
// MI455X (gfx1250) — matrix operand order repaired and checked by reading, not compiled, not run on hardware
//
#include <hip/hip_runtime.h>

typedef _Float16 v8h  __attribute__((ext_vector_type(8)));
typedef _Float16 v16h __attribute__((ext_vector_type(16)));
typedef float    v8f  __attribute__((ext_vector_type(8)));

union F16Frag { v16h v; v8h h[2]; };

#define N_RES 384
#define CZ    128
#define NROWS (N_RES * N_RES)   // 147456

static __device__ __forceinline__ v8f wmma16x16x32(v16h a, v16h b, v8f c) {
  // D = A(16x32 f16) * B(32x16 f16) + C(16x16 f32)
  return __builtin_amdgcn_wmma_f32_16x16x32_f16(false, a, false, b, (short)0, c,
                                                false, false);
}

static __device__ __forceinline__ float sigmoidf_(float x) {
  return 1.0f / (1.0f + __expf(-x));
}

// ---------------------------------------------------------------------------
// Kernel 0: convert the six 128x128 weight matrices to fp16 (order:
// a_g, a_p, b_g, b_p, g_w, z_w)
// ---------------------------------------------------------------------------
__global__ __launch_bounds__(256) void k_wcvt(
    const float* __restrict__ ag, const float* __restrict__ ap,
    const float* __restrict__ bg, const float* __restrict__ bp,
    const float* __restrict__ gw, const float* __restrict__ zw,
    _Float16* __restrict__ wf) {
  int i = blockIdx.x * 256 + threadIdx.x;    // 6*16384 total
  int which = i >> 14, e = i & 16383;
  const float* s = (which == 0) ? ag : (which == 1) ? ap : (which == 2) ? bg
                 : (which == 3) ? bp : (which == 4) ? gw : zw;
  wf[i] = (_Float16)s[e];
}

// ---------------------------------------------------------------------------
// Kernel 1: fused LN(z) + 5 projections via WMMA.
//   a = mask * sigmoid(zn@agW^T + agB) * (zn@apW^T + apB)  -> aT[h][row] fp16
//   b = likewise                                            -> bT[h][row] fp16
//   gate = sigmoid(zn@gW^T + gB)                            -> gate[row][o] fp16
// Block = 256 thr (8 waves), 64 rows per block.
// ---------------------------------------------------------------------------
__global__ __launch_bounds__(256) void k_proj(
    const float* __restrict__ z, const float* __restrict__ mask,
    const float* __restrict__ lnw, const float* __restrict__ lnb,
    const _Float16* __restrict__ wf,
    const float* __restrict__ agb, const float* __restrict__ apb,
    const float* __restrict__ bgb, const float* __restrict__ bpb,
    const float* __restrict__ gbb,
    _Float16* __restrict__ aT, _Float16* __restrict__ bT,
    _Float16* __restrict__ gate) {
  __shared__ _Float16 zn[64 * 136];       // LN'd activations, fp16
  __shared__ _Float16 stg[2][128 * 72];   // transposed a/b staging [h][row]
  const int tid = threadIdx.x;
  const int rowbase = blockIdx.x * 64;

  { // ---- LayerNorm: 4 threads per row, 32 channels each ----
    const int r = tid >> 2, p = tid & 3;
    const float* zr = z + (size_t)(rowbase + r) * CZ + p * 32;
    float4 v[8];
    float s = 0.f, ss = 0.f;
#pragma unroll
    for (int u = 0; u < 8; ++u) {
      v[u] = ((const float4*)zr)[u];
      s  += v[u].x + v[u].y + v[u].z + v[u].w;
      ss += v[u].x * v[u].x + v[u].y * v[u].y + v[u].z * v[u].z + v[u].w * v[u].w;
    }
    s  += __shfl_xor(s, 1);  s  += __shfl_xor(s, 2);
    ss += __shfl_xor(ss, 1); ss += __shfl_xor(ss, 2);
    const float mu   = s * (1.f / 128.f);
    const float rstd = rsqrtf(ss * (1.f / 128.f) - mu * mu + 1e-5f);
    _Float16* dst = &zn[r * 136 + p * 32];
#pragma unroll
    for (int u = 0; u < 8; ++u) {
      const int c = p * 32 + u * 4;
      dst[u * 4 + 0] = (_Float16)((v[u].x - mu) * rstd * lnw[c + 0] + lnb[c + 0]);
      dst[u * 4 + 1] = (_Float16)((v[u].y - mu) * rstd * lnw[c + 1] + lnb[c + 1]);
      dst[u * 4 + 2] = (_Float16)((v[u].z - mu) * rstd * lnw[c + 2] + lnb[c + 2]);
      dst[u * 4 + 3] = (_Float16)((v[u].w - mu) * rstd * lnw[c + 3] + lnb[c + 3]);
    }
  }
  __syncthreads();

  // Wave id forced scalar so job decode / weight pointers are SALU.
  const int wvs = __builtin_amdgcn_readfirstlane(tid >> 5);
  const int l = tid & 31, m = l & 15, kh = l >> 4;

  // ---- (a,b) pair jobs: 2 pairs x 4 m-tiles x 8 n-tiles = 64 tiles / 8 waves
  // All 12 fragment loads issued up front -> one wait, 8 back-to-back WMMAs.
  for (int job = wvs; job < 64; job += 8) {
    const int pair = job >> 5;                      // scalar
    const int mt = (job & 31) >> 3, nt = job & 7;   // scalar
    const _Float16* Wg = wf + pair * 2 * 16384;
    const _Float16* Wp = Wg + 16384;
    F16Frag af[4], bf[4], pf[4];
#pragma unroll
    for (int ks = 0; ks < 4; ++ks) {
      const int k0 = ks * 32;
      af[ks].h[0] = *(const v8h*)&zn[(mt * 16 + m) * 136 + k0 + 16 * kh];
      af[ks].h[1] = *(const v8h*)&zn[(mt * 16 + m) * 136 + k0 + 8 + 16 * kh];
      const _Float16* wr = Wg + (nt * 16 + m) * 128 + k0 + 16 * kh;
      bf[ks].h[0] = *(const v8h*)wr;
      bf[ks].h[1] = *(const v8h*)(wr + 8);
      const _Float16* pr = Wp + (nt * 16 + m) * 128 + k0 + 16 * kh;
      pf[ks].h[0] = *(const v8h*)pr;
      pf[ks].h[1] = *(const v8h*)(pr + 8);
    }
    v8f accg = {};
    v8f accp = {};
#pragma unroll
    for (int ks = 0; ks < 4; ++ks) {
      accg = wmma16x16x32(af[ks].v, bf[ks].v, accg);
      accp = wmma16x16x32(af[ks].v, pf[ks].v, accp);
    }
    const int o = nt * 16 + m;  // C/D frag: N = lane&15
    const float bg = (pair == 0) ? agb[o] : bgb[o];
    const float bp = (pair == 0) ? apb[o] : bpb[o];
    _Float16* st = stg[pair];
#pragma unroll
    for (int rr = 0; rr < 8; ++rr) {
      const int rowl = mt * 16 + rr + 8 * kh;  // M = r + 8*(lane>=16)
      const float mk = mask[rowbase + rowl];
      st[o * 72 + rowl] =
          (_Float16)(mk * sigmoidf_(accg[rr] + bg) * (accp[rr] + bp));
    }
  }

  // ---- gate jobs: 4 m-tiles x 8 n-tiles = 32 tiles / 8 waves
  {
    const _Float16* Wg = wf + 4 * 16384;
    for (int job = wvs; job < 32; job += 8) {
      const int mt = job >> 3, nt = job & 7;   // scalar
      F16Frag af[4], bf[4];
#pragma unroll
      for (int ks = 0; ks < 4; ++ks) {
        const int k0 = ks * 32;
        af[ks].h[0] = *(const v8h*)&zn[(mt * 16 + m) * 136 + k0 + 16 * kh];
        af[ks].h[1] = *(const v8h*)&zn[(mt * 16 + m) * 136 + k0 + 8 + 16 * kh];
        const _Float16* wr = Wg + (nt * 16 + m) * 128 + k0 + 16 * kh;
        bf[ks].h[0] = *(const v8h*)wr;
        bf[ks].h[1] = *(const v8h*)(wr + 8);
      }
      v8f accg = {};
#pragma unroll
      for (int ks = 0; ks < 4; ++ks)
        accg = wmma16x16x32(af[ks].v, bf[ks].v, accg);
      const int o = nt * 16 + m;
      const float bg = gbb[o];
#pragma unroll
      for (int rr = 0; rr < 8; ++rr) {
        const int rowl = mt * 16 + rr + 8 * kh;
        gate[(size_t)(rowbase + rowl) * CZ + o] =
            (_Float16)sigmoidf_(accg[rr] + bg);
      }
    }
  }
  __syncthreads();

  { // ---- coalesced channel-major writeout: 64B contiguous per thread ----
    const int h = tid >> 1, half = tid & 1;
    const _Float16* sa = &stg[0][h * 72 + half * 32];
    const _Float16* sb = &stg[1][h * 72 + half * 32];
    _Float16* da = aT + (size_t)h * NROWS + rowbase + half * 32;
    _Float16* db = bT + (size_t)h * NROWS + rowbase + half * 32;
#pragma unroll
    for (int u = 0; u < 32; u += 8) {
      *(v8h*)(da + u) = *(const v8h*)(sa + u);
      *(v8h*)(db + u) = *(const v8h*)(sb + u);
    }
  }
}

// ---------------------------------------------------------------------------
// Kernel 2: triangle contraction. For each channel h: X_h = A_h * B_h^T,
// all 384x384 f16 GEMMs. Block = 192 thr (6 waves); A panel (32x384) staged
// once in LDS and shared by all 6 waves; each wave owns a 32x64 output tile
// (2x4 accumulators). Fragments are register double-buffered so WMMA of step
// s overlaps the global/LDS loads of step s+1.
// ---------------------------------------------------------------------------
__global__ __launch_bounds__(192) void k_tri(const _Float16* __restrict__ aT,
                                             const _Float16* __restrict__ bT,
                                             float* __restrict__ xT) {
  __shared__ _Float16 As[32 * 392];   // 32 rows x 384 fp16 (+8 pad per row)
  const int h  = blockIdx.y;
  const int i0 = blockIdx.x * 32;
  const int tid = threadIdx.x;
  const _Float16* Ah = aT + (size_t)h * NROWS;
  const _Float16* Bh = bT + (size_t)h * NROWS;

  // cooperative A-panel stage: 1536 16B chunks over 192 threads (8 each)
#pragma unroll
  for (int u = 0; u < 8; ++u) {
    const int c = tid + 192 * u;
    const int row = c / 48, col = (c % 48) * 8;
    *(v8h*)&As[row * 392 + col] = *(const v8h*)&Ah[(i0 + row) * N_RES + col];
  }
  __syncthreads();

  const int wvs = __builtin_amdgcn_readfirstlane(tid >> 5);  // scalar wave id
  const int l = tid & 31, m = l & 15, kh = l >> 4;
  const int j0 = wvs * 64;

  v8f acc[2][4] = {};
  F16Frag af[2][2];
  F16Frag bf[2][4];

  auto loadA = [&](int buf, int k0) {
#pragma unroll
    for (int mi = 0; mi < 2; ++mi) {
      const _Float16* ar = &As[(mi * 16 + m) * 392 + k0];
      af[buf][mi].h[0] = *(const v8h*)(ar + 16 * kh);
      af[buf][mi].h[1] = *(const v8h*)(ar + 8 + 16 * kh);
    }
  };
  auto loadB = [&](int buf, int k0) {
#pragma unroll
    for (int ni = 0; ni < 4; ++ni) {
      const _Float16* br = Bh + (j0 + ni * 16 + m) * N_RES + k0 + 16 * kh;
      bf[buf][ni].h[0] = *(const v8h*)br;
      bf[buf][ni].h[1] = *(const v8h*)(br + 8);
    }
  };

  loadA(0, 0);
  loadB(0, 0);
#pragma unroll
  for (int s = 0; s < 12; ++s) {
    const int cur = s & 1, nxt = cur ^ 1;
    if (s < 11) {
      loadA(nxt, (s + 1) * 32);
      loadB(nxt, (s + 1) * 32);
      if (s + 2 < 12)  // gfx1250 global_prefetch_b8, two K-panels ahead
        __builtin_prefetch(Bh + (j0 + m) * N_RES + (s + 2) * 32, 0, 1);
    }
#pragma unroll
    for (int mi = 0; mi < 2; ++mi)
#pragma unroll
      for (int ni = 0; ni < 4; ++ni)
        acc[mi][ni] = wmma16x16x32(af[cur][mi].v, bf[cur][ni].v, acc[mi][ni]);
  }

  float* Xh = xT + (size_t)h * NROWS;
#pragma unroll
  for (int mi = 0; mi < 2; ++mi)
#pragma unroll
    for (int ni = 0; ni < 4; ++ni)
#pragma unroll
      for (int rr = 0; rr < 8; ++rr)
        Xh[(i0 + mi * 16 + rr + 8 * kh) * N_RES + (j0 + ni * 16 + m)] =
            acc[mi][ni][rr];
}

// ---------------------------------------------------------------------------
// Kernel 3: out = sigmoid-gate * (LN_h(x) @ z_w^T + z_b).  64 rows / block.
// ---------------------------------------------------------------------------
__global__ __launch_bounds__(256) void k_out(
    const float* __restrict__ xT, const _Float16* __restrict__ gate,
    const _Float16* __restrict__ wz, const float* __restrict__ lnw,
    const float* __restrict__ lnb, const float* __restrict__ zb,
    float* __restrict__ out) {
  __shared__ float    xs[64 * 132];
  __shared__ _Float16 xn[64 * 136];
  const int tid = threadIdx.x;
  const int rowbase = blockIdx.x * 64;
  { // gather channel-major x (coalesced) -> row-major LDS
    const int h = tid >> 1, half = tid & 1;
    const float* src = xT + (size_t)h * NROWS + rowbase + half * 32;
#pragma unroll
    for (int j = 0; j < 32; ++j) xs[(half * 32 + j) * 132 + h] = src[j];
  }
  __syncthreads();
  { // LayerNorm over h
    const int r = tid >> 2, p = tid & 3;
    const float* xr = &xs[r * 132 + p * 32];
    float s = 0.f, ss = 0.f;
    float buf[32];
#pragma unroll
    for (int c = 0; c < 32; ++c) {
      buf[c] = xr[c];
      s += buf[c];
      ss += buf[c] * buf[c];
    }
    s  += __shfl_xor(s, 1);  s  += __shfl_xor(s, 2);
    ss += __shfl_xor(ss, 1); ss += __shfl_xor(ss, 2);
    const float mu   = s * (1.f / 128.f);
    const float rstd = rsqrtf(ss * (1.f / 128.f) - mu * mu + 1e-5f);
    _Float16* dst = &xn[r * 136 + p * 32];
#pragma unroll
    for (int c = 0; c < 32; ++c)
      dst[c] = (_Float16)((buf[c] - mu) * rstd * lnw[p * 32 + c] + lnb[p * 32 + c]);
  }
  __syncthreads();
  const int wvs = __builtin_amdgcn_readfirstlane(tid >> 5);  // scalar wave id
  const int l = tid & 31, m = l & 15, kh = l >> 4;
  for (int job = wvs; job < 32; job += 8) {
    const int mt = job >> 3, nt = job & 7;   // scalar
    F16Frag af[4], bf[4];
#pragma unroll
    for (int ks = 0; ks < 4; ++ks) {
      const int k0 = ks * 32;
      af[ks].h[0] = *(const v8h*)&xn[(mt * 16 + m) * 136 + k0 + 16 * kh];
      af[ks].h[1] = *(const v8h*)&xn[(mt * 16 + m) * 136 + k0 + 8 + 16 * kh];
      const _Float16* wr = wz + (nt * 16 + m) * 128 + k0 + 16 * kh;
      bf[ks].h[0] = *(const v8h*)wr;
      bf[ks].h[1] = *(const v8h*)(wr + 8);
    }
    v8f acc = {};
#pragma unroll
    for (int ks = 0; ks < 4; ++ks)
      acc = wmma16x16x32(af[ks].v, bf[ks].v, acc);
    const int o = nt * 16 + m;
    const float bz = zb[o];
#pragma unroll
    for (int rr = 0; rr < 8; ++rr) {
      const size_t g = (size_t)(rowbase + mt * 16 + rr + 8 * kh) * CZ + o;
      out[g] = (float)gate[g] * (acc[rr] + bz);
    }
  }
}

// ---------------------------------------------------------------------------
extern "C" void kernel_launch(void* const* d_in, const int* in_sizes, int n_in,
                              void* d_out, int out_size, void* d_ws,
                              size_t ws_size, hipStream_t stream) {
  const float* z        = (const float*)d_in[0];
  const float* mask     = (const float*)d_in[1];
  const float* ln_in_w  = (const float*)d_in[2];
  const float* ln_in_b  = (const float*)d_in[3];
  const float* a_g_w    = (const float*)d_in[4];
  const float* a_g_b    = (const float*)d_in[5];
  const float* a_p_w    = (const float*)d_in[6];
  const float* a_p_b    = (const float*)d_in[7];
  const float* b_g_w    = (const float*)d_in[8];
  const float* b_g_b    = (const float*)d_in[9];
  const float* b_p_w    = (const float*)d_in[10];
  const float* b_p_b    = (const float*)d_in[11];
  const float* g_w      = (const float*)d_in[12];
  const float* g_b      = (const float*)d_in[13];
  const float* ln_out_w = (const float*)d_in[14];
  const float* ln_out_b = (const float*)d_in[15];
  const float* z_w      = (const float*)d_in[16];
  const float* z_b      = (const float*)d_in[17];

  char* ws = (char*)d_ws;
  const size_t HALF_MAT = (size_t)128 * NROWS * sizeof(_Float16);  // 37748736
  _Float16* wf   = (_Float16*)(ws);                         // 6*16384 fp16
  _Float16* aT   = (_Float16*)(ws + 196608);                // [h][row] fp16
  _Float16* bT   = (_Float16*)(ws + 196608 + HALF_MAT);     // [h][row] fp16
  _Float16* gate = (_Float16*)(ws + 196608 + 2 * HALF_MAT); // [row][o] fp16
  float*    xT   = (float*)   (ws + 196608 + 3 * HALF_MAT); // [h][row] f32
  float*    outp = (float*)d_out;

  k_wcvt<<<384, 256, 0, stream>>>(a_g_w, a_p_w, b_g_w, b_p_w, g_w, z_w, wf);
  k_proj<<<NROWS / 64, 256, 0, stream>>>(z, mask, ln_in_w, ln_in_b, wf, a_g_b,
                                         a_p_b, b_g_b, b_p_b, g_b, aT, bT, gate);
  k_tri<<<dim3(N_RES / 32, 128), 192, 0, stream>>>(aT, bT, xT);
  k_out<<<NROWS / 64, 256, 0, stream>>>(xT, gate, wf + 5 * 16384, ln_out_w,
                                        ln_out_b, z_b, outp);
}
